// MyEdgeConvBlock_19473381720811
// MI455X (gfx1250) — compile-verified, matrix operand order checked by reading, not run on hardware
//
#include <hip/hip_runtime.h>
#include <hip/hip_fp16.h>

typedef _Float16 half8  __attribute__((ext_vector_type(8)));
typedef _Float16 half16 __attribute__((ext_vector_type(16)));
typedef float    float8 __attribute__((ext_vector_type(8)));

#define BN_EPS 1e-5f

// workspace byte offsets
#define WS_STATS 0        // sum[64], sumsq[64] f32
#define WS_SCALE 512      // 64 f32
#define WS_SHIFT 768      // 64 f32
#define WS_W1T   4096     // 128x128 f16 (W1 transposed, [N][K]) -- contiguous with W2T
#define WS_W2T   36864    // 64x128 f16  (W2 transposed, [N][K])
#define WS_XN    65536    // n_nodes*64 f16 normalized node features

// ---------------------------------------------------------------- utilities
__device__ __forceinline__ half8 ld8(const _Float16* p) {
    return *(const half8*)p;
}

__device__ __forceinline__ half16 cat16(half8 lo, half8 hi) {
    half16 r;
#pragma unroll
    for (int i = 0; i < 8; ++i) { r[i] = lo[i]; r[i + 8] = hi[i]; }
    return r;
}

__device__ __forceinline__ void atomic_max_f32(float* addr, float val) {
    unsigned long long a = (unsigned long long)addr;
    asm volatile("global_atomic_max_num_f32 %0, %1, off"
                 :: "v"(a), "v"(val) : "memory");
}

// async global -> LDS DMA (bypasses VGPRs, tracked by ASYNCcnt)
__device__ __forceinline__ void async_g2l_b128(unsigned lds_off, const void* gaddr) {
    unsigned long long ga = (unsigned long long)gaddr;
    asm volatile("global_load_async_to_lds_b128 %0, %1, off"
                 :: "v"(lds_off), "v"(ga) : "memory");
}

// ---------------------------------------------------------------- kernels
__global__ void k_zero(float* out, int n, float* stats) {
    int i = blockIdx.x * blockDim.x + threadIdx.x;
    int stride = gridDim.x * blockDim.x;
    for (; i < n; i += stride) out[i] = 0.0f;
    if (blockIdx.x == 0 && threadIdx.x < 128) stats[threadIdx.x] = 0.0f;
}

// per-channel sum / sum-of-squares (coalesced: 64 consecutive channels per row)
__global__ void k_stats(const float* __restrict__ x, int n_nodes, float* stats) {
    int t = threadIdx.x;
    int ch = t & 63;
    int g = t >> 6;  // 0..3 node groups per block
    float s = 0.0f, s2 = 0.0f;
    for (int node = blockIdx.x * 4 + g; node < n_nodes; node += gridDim.x * 4) {
        float v = x[node * 64 + ch];
        s += v; s2 += v * v;
    }
    __shared__ float ls[256], ls2[256];
    ls[t] = s; ls2[t] = s2;
    __syncthreads();
    if (t < 64) {
        float ts  = ls[t]  + ls[t + 64]  + ls[t + 128]  + ls[t + 192];
        float ts2 = ls2[t] + ls2[t + 64] + ls2[t + 128] + ls2[t + 192];
        atomicAdd(&stats[t], ts);
        atomicAdd(&stats[64 + t], ts2);
    }
}

__global__ void k_finalize(const float* __restrict__ stats,
                           const float* __restrict__ gamma,
                           const float* __restrict__ beta,
                           int n_nodes, float* scale, float* shift) {
    int c = threadIdx.x;
    if (c < 64) {
        float inv = 1.0f / (float)n_nodes;
        float mean = stats[c] * inv;
        float var = stats[64 + c] * inv - mean * mean;
        float sc = gamma[c] * rsqrtf(var + BN_EPS);
        scale[c] = sc;
        shift[c] = beta[c] - mean * sc;
    }
}

__global__ void k_normalize(const float* __restrict__ x,
                            const float* __restrict__ scale,
                            const float* __restrict__ shift,
                            _Float16* __restrict__ xn, int total) {
    int i = blockIdx.x * blockDim.x + threadIdx.x;
    int stride = gridDim.x * blockDim.x;
    for (; i < total; i += stride) {
        int c = i & 63;
        xn[i] = (_Float16)(x[i] * scale[c] + shift[c]);
    }
}

// W1[k][n] (128x128) -> w1t[n][k] f16 ; W2[k][n] (128x64) -> w2t[n][k] f16
__global__ void k_weights(const float* __restrict__ W1, const float* __restrict__ W2,
                          _Float16* __restrict__ w1t, _Float16* __restrict__ w2t) {
    int i = blockIdx.x * blockDim.x + threadIdx.x;
    int stride = gridDim.x * blockDim.x;
    for (int j = i; j < 128 * 128; j += stride) {
        int n = j >> 7, k = j & 127;
        w1t[j] = (_Float16)W1[k * 128 + n];
    }
    for (int j = i; j < 64 * 128; j += stride) {
        int n = j >> 7, k = j & 127;
        w2t[j] = (_Float16)W2[k * 64 + n];
    }
}

// ---------------------------------------------------------------- edge MLP
// One wave32 handles 16 edges: feat[16x128] -> relu(@W1+b1)[16x128] -> (@W2+b2)[16x64]
// -> global_atomic_max_num_f32 scatter to out[dst].
// Weights staged once per block into LDS via async global->LDS DMA; all WMMA
// B-fragments then come from ds_load_b128, freeing the vmem pipe for the
// random node gathers + atomics that actually hit L2/HBM.
__global__ void __launch_bounds__(256)
k_edgeconv(const _Float16* __restrict__ xn, const int* __restrict__ ei, long n_edges,
           const _Float16* __restrict__ wts,   // W1t (16384 halves) ++ W2t (8192 halves)
           const float* __restrict__ b1, const float* __restrict__ b2,
           float* __restrict__ out) {
    __shared__ _Float16 WL[24576];        // 48 KB: W1t then W2t
    __shared__ _Float16 Hl[8][16 * 128];  // 32 KB: per-wave ReLU'd hidden tiles

    // ---- stage weights: 3072 x 16B chunks, async DMA straight into LDS
    {
        unsigned lbase = (unsigned)(unsigned long long)(void*)&WL[0];
        const char* g = (const char*)wts;
#pragma unroll
        for (int c = threadIdx.x; c < 3072; c += 256) {
            async_g2l_b128(lbase + c * 16, g + c * 16);
        }
        asm volatile("s_wait_asynccnt 0" ::: "memory");
        __syncthreads();
    }
    const _Float16* w1l = &WL[0];
    const _Float16* w2l = &WL[16384];

    const int lane = threadIdx.x & 31;
    const int wave = threadIdx.x >> 5;
    const int eRow = lane & 15;   // matrix row M owned by this lane (A layout)
    const int h    = lane >> 4;   // half-wave select (K-group split)
    const int kb   = h * 8;       // K base within 32-wide K step
    const int col  = lane & 15;   // N column for C/D layout

    long tile = (long)blockIdx.x * 8 + wave;
    long e = tile * 16 + eRow;
    if (e >= n_edges) e = n_edges - 1;   // duplicate-edge clamp: max is idempotent

    const int s = ei[e];             // source j
    const int d = ei[n_edges + e];   // target i

    const _Float16* xi = xn + (long)d * 64;
    const _Float16* xj = xn + (long)s * 64;

    // gather both node rows as A-layout fragments (2x16B per K-step per lane)
    half8 di0 = ld8(xi + kb),      di1 = ld8(xi + kb + 16);
    half8 di2 = ld8(xi + 32 + kb), di3 = ld8(xi + 32 + kb + 16);
    half8 sj0 = ld8(xj + kb),      sj1 = ld8(xj + kb + 16);
    half8 sj2 = ld8(xj + 32 + kb), sj3 = ld8(xj + 32 + kb + 16);

    half16 A[4];
    A[0] = cat16(di0, di1);              // feat[0:32]   = x_i[0:32]
    A[1] = cat16(di2, di3);              // feat[32:64]  = x_i[32:64]
    A[2] = cat16(sj0 - di0, sj1 - di1);  // feat[64:96]  = (x_j-x_i)[0:32]
    A[3] = cat16(sj2 - di2, sj3 - di3);  // feat[96:128] = (x_j-x_i)[32:64]

    _Float16* hrow = &Hl[wave][0];

    // ---- GEMM1: H = relu(feat @ W1 + b1), 8 N-tiles x 4 K-steps = 32 WMMA
#pragma unroll
    for (int n = 0; n < 8; ++n) {
        float bias = b1[n * 16 + col];
        float8 acc;
#pragma unroll
        for (int v = 0; v < 8; ++v) acc[v] = bias;
        const _Float16* wb = w1l + (n * 16 + col) * 128;
#pragma unroll
        for (int k = 0; k < 4; ++k) {
            half16 B = cat16(ld8(wb + k * 32 + kb), ld8(wb + k * 32 + kb + 16));
            acc = __builtin_amdgcn_wmma_f32_16x16x32_f16(
                false, A[k], false, B, (short)0, acc, false, false);
        }
        // ReLU + store to LDS for C-layout -> A-layout change
#pragma unroll
        for (int v = 0; v < 8; ++v) {
            float r = acc[v] > 0.0f ? acc[v] : 0.0f;
            hrow[(v + 8 * h) * 128 + n * 16 + col] = (_Float16)r;
        }
    }

    // re-load H as A-layout fragments (same-wave LDS is in order)
    half16 A2[4];
#pragma unroll
    for (int k = 0; k < 4; ++k) {
        const _Float16* hp = hrow + eRow * 128 + k * 32 + kb;
        A2[k] = cat16(*(const half8*)hp, *(const half8*)(hp + 16));
    }

    // ---- GEMM2: msg = H @ W2 + b2, 4 N-tiles x 4 K-steps = 16 WMMA
#pragma unroll
    for (int n2 = 0; n2 < 4; ++n2) {
        float bias = b2[n2 * 16 + col];
        float8 acc;
#pragma unroll
        for (int v = 0; v < 8; ++v) acc[v] = bias;
        const _Float16* wb = w2l + (n2 * 16 + col) * 128;
#pragma unroll
        for (int k = 0; k < 4; ++k) {
            half16 B = cat16(ld8(wb + k * 32 + kb), ld8(wb + k * 32 + kb + 16));
            acc = __builtin_amdgcn_wmma_f32_16x16x32_f16(
                false, A2[k], false, B, (short)0, acc, false, false);
        }
        // scatter-max: out zero-initialized, so max(0,.) == trailing ReLU,
        // and empty segments stay 0 (== where(isfinite) fixup)
#pragma unroll
        for (int v = 0; v < 8; ++v) {
            int drow = __shfl(d, v + 8 * h, 32);
            atomic_max_f32(out + (long)drow * 64 + n2 * 16 + col, acc[v]);
        }
    }
}

// ---------------------------------------------------------------- launch
extern "C" void kernel_launch(void* const* d_in, const int* in_sizes, int n_in,
                              void* d_out, int out_size, void* d_ws, size_t ws_size,
                              hipStream_t stream) {
    const float* x     = (const float*)d_in[0];
    const int*   ei    = (const int*)d_in[1];
    const float* gamma = (const float*)d_in[2];
    const float* beta  = (const float*)d_in[3];
    const float* W1    = (const float*)d_in[4];
    const float* b1    = (const float*)d_in[5];
    const float* W2    = (const float*)d_in[6];
    const float* b2    = (const float*)d_in[7];

    int n_nodes  = in_sizes[0] / 64;
    long n_edges = (long)in_sizes[1] / 2;

    char* ws = (char*)d_ws;
    float*     stats = (float*)(ws + WS_STATS);
    float*     scale = (float*)(ws + WS_SCALE);
    float*     shift = (float*)(ws + WS_SHIFT);
    _Float16*  w1t   = (_Float16*)(ws + WS_W1T);
    _Float16*  w2t   = (_Float16*)(ws + WS_W2T);
    _Float16*  xnh   = (_Float16*)(ws + WS_XN);

    k_zero<<<512, 256, 0, stream>>>((float*)d_out, out_size, stats);
    k_stats<<<256, 256, 0, stream>>>(x, n_nodes, stats);
    k_finalize<<<1, 64, 0, stream>>>(stats, gamma, beta, n_nodes, scale, shift);
    k_normalize<<<1024, 256, 0, stream>>>(x, scale, shift, xnh, n_nodes * 64);
    k_weights<<<64, 256, 0, stream>>>(W1, W2, w1t, w2t);

    long tiles = (n_edges + 15) / 16;
    int blocks = (int)((tiles + 7) / 8);
    k_edgeconv<<<blocks, 256, 0, stream>>>(xnh, ei, n_edges, w1t, b1, b2,
                                           (float*)d_out);
}
